// GPT_55044300865697
// MI455X (gfx1250) — matrix operand order repaired and checked by reading, not compiled, not run on hardware
//
#include <hip/hip_runtime.h>

// ---------------- types ----------------
typedef _Float16 h16_t;
typedef __attribute__((ext_vector_type(2)))  _Float16 v2h;
typedef __attribute__((ext_vector_type(4)))  _Float16 v4h;
typedef __attribute__((ext_vector_type(8)))  _Float16 v8h;
typedef __attribute__((ext_vector_type(16))) _Float16 v16h;
typedef __attribute__((ext_vector_type(4)))  float    v4f;
typedef __attribute__((ext_vector_type(8)))  float    v8f;
typedef __attribute__((ext_vector_type(2)))  int      v2i_t;

union Frag16 {
  v16h v;
  v8h  h[2];
  v4h  q[4];
};

// ---------------- model constants ----------------
#define BB   2
#define TT   1024
#define CC   768
#define HH   12
#define HD   64
#define LLAY 12
#define VV   50257
#define MM   (BB*TT)          // 2048 tokens

// ---------------- CDNA5 async-to-LDS support ----------------
#if defined(__AMDGCN__) && __has_builtin(__builtin_amdgcn_global_load_async_to_lds_b64)
#define HAVE_ASYNC_LDS 1
#else
#define HAVE_ASYNC_LDS 0
#endif

// cast helpers: builtin wants (const v2i addrspace(1)*, v2i addrspace(3)*, imm, imm)
#define GAS_V2I(p) ((__attribute__((address_space(1))) v2i_t*)(void*)(p))
#define LAS_V2I(p) ((__attribute__((address_space(3))) v2i_t*)(void*)(p))

__device__ __forceinline__ void wait_async0() {
#if defined(__AMDGCN__)
#if __has_builtin(__builtin_amdgcn_s_wait_asynccnt)
  __builtin_amdgcn_s_wait_asynccnt(0);
#else
  asm volatile("s_wait_asynccnt 0x0" ::: "memory");
#endif
#endif
}

__device__ __forceinline__ v8f wmma_f16(v16h a, v16h b, v8f c) {
  return __builtin_amdgcn_wmma_f32_16x16x32_f16(false, a, false, b, (short)0, c, false, false);
}

__device__ __forceinline__ float gelu_f(float x) {
  const float k = 0.7978845608028654f;   // sqrt(2/pi)
  float x3 = x * x * x;
  return 0.5f * x * (1.0f + tanhf(k * (x + 0.044715f * x3)));
}

// ---------------- embedding: x = wte[idx] + wpe ----------------
__global__ __launch_bounds__(256) void k_embed(const int* __restrict__ idx,
                                               const float* __restrict__ wte,
                                               const float* __restrict__ wpe,
                                               float* __restrict__ xf) {
  int i = blockIdx.x * 256 + threadIdx.x;         // over MM*CC
  if (i >= MM * CC) return;
  int row = i / CC, c = i % CC;
  int tpos = row % TT;
  xf[i] = wte[(long long)idx[row] * CC + c] + wpe[tpos * CC + c];
}

// ---------------- layernorm -> f16 ----------------
__global__ __launch_bounds__(256) void k_ln(const float* __restrict__ x,
                                            const float* __restrict__ w,
                                            const float* __restrict__ bsrc,
                                            h16_t* __restrict__ out) {
  __shared__ float red[256];
  const int row = blockIdx.x;
  const float* xr = x + (size_t)row * CC;
  float s = 0.f;
  for (int c = threadIdx.x; c < CC; c += 256) s += xr[c];
  red[threadIdx.x] = s; __syncthreads();
  for (int st = 128; st > 0; st >>= 1) {
    if (threadIdx.x < st) red[threadIdx.x] += red[threadIdx.x + st];
    __syncthreads();
  }
  float mean = red[0] * (1.0f / (float)CC);
  __syncthreads();
  float vs = 0.f;
  for (int c = threadIdx.x; c < CC; c += 256) { float d = xr[c] - mean; vs += d * d; }
  red[threadIdx.x] = vs; __syncthreads();
  for (int st = 128; st > 0; st >>= 1) {
    if (threadIdx.x < st) red[threadIdx.x] += red[threadIdx.x + st];
    __syncthreads();
  }
  float inv = rsqrtf(red[0] * (1.0f / (float)CC) + 1e-5f);
  for (int c = threadIdx.x; c < CC; c += 256)
    out[(size_t)row * CC + c] = (h16_t)((xr[c] - mean) * inv * w[c] + bsrc[c]);
}

// ------- qkv pack: f32 [M,3C] -> f16 q/k [b,h,t,hd], v transposed [b,h,hd,t] -------
__global__ __launch_bounds__(256) void k_pack(const float* __restrict__ qkv,
                                              h16_t* __restrict__ q16,
                                              h16_t* __restrict__ k16,
                                              h16_t* __restrict__ vT16) {
  int i = blockIdx.x * 256 + threadIdx.x;         // over MM*3CC
  if (i >= MM * 3 * CC) return;
  int row = i / (3 * CC), c = i % (3 * CC);
  int which = c / CC, cc = c % CC;
  int h = cc / HD, hd = cc % HD;
  int b = row / TT, t = row % TT;
  h16_t v = (h16_t)qkv[i];
  if (which == 0) {
    q16[((size_t)(b * HH + h) * TT + t) * HD + hd] = v;
  } else if (which == 1) {
    k16[((size_t)(b * HH + h) * TT + t) * HD + hd] = v;
  } else {
    vT16[((size_t)(b * HH + h) * HD + hd) * TT + t] = v;   // transposed per head
  }
}

// ---------------- generic WMMA GEMM, double-buffered pipeline ----------------
// C[M,N] = A(f16,[M,K]) @ B(f32->f16,[K,N]) (+bias)(+res)(gelu?) -> outF/outH
#define BM 128
#define BN 128
#define BK 32
#define SA_STR 36     // halfs; 72B row stride (8B aligned chunks)
#define BKSTR  40     // halfs; 80B row stride (16B aligned chunks), transposed B

// A tile: f16 copy. Async-to-LDS when available, register-staged otherwise.
__device__ __forceinline__ void copyA_tile(const h16_t* __restrict__ A, h16_t* dst,
                                           int tid, int blockM, int K, int k0) {
#if HAVE_ASYNC_LDS
  #pragma unroll
  for (int l = 0; l < 4; l++) {
    int ix = l * 256 + tid;
    int r  = ix >> 3;                  // 8 chunks of 4 halfs per row
    int cq = (ix & 7) * 4;
    const h16_t* src = A + (long long)(blockM + r) * K + k0 + cq;
    __builtin_amdgcn_global_load_async_to_lds_b64(GAS_V2I(src),
                                                  LAS_V2I(dst + r * SA_STR + cq), 0, 0);
  }
#else
  v4h t[4];
  #pragma unroll
  for (int l = 0; l < 4; l++) {
    int ix = l * 256 + tid;
    int r  = ix >> 3;
    int cq = (ix & 7) * 4;
    t[l] = *(const v4h*)(A + (long long)(blockM + r) * K + k0 + cq);
  }
  #pragma unroll
  for (int l = 0; l < 4; l++) {
    int ix = l * 256 + tid;
    int r  = ix >> 3;
    int cq = (ix & 7) * 4;
    *(v4h*)(dst + r * SA_STR + cq) = t[l];
  }
#endif
}

// B tile: f32 global loads -> registers (issued early, no waits interleaved).
__device__ __forceinline__ void loadB_regs(const float* __restrict__ Bw, int tid,
                                           int blockN, int N, int k0,
                                           float* t0, float* t1) {
  #pragma unroll
  for (int l = 0; l < 8; l++) {
    int ix = l * 256 + tid;            // 2048 K-pair chunks
    int col = ix & 127;
    int kp  = ix >> 7;                 // 0..15
    int gcol = blockN + col;
    float a = 0.f, b = 0.f;
    if (gcol < N) {
      a = Bw[(long long)(k0 + 2 * kp) * N + gcol];
      b = Bw[(long long)(k0 + 2 * kp + 1) * N + gcol];
    }
    t0[l] = a; t1[l] = b;
  }
}

__device__ __forceinline__ void storeB_lds(h16_t* dst, int tid,
                                           const float* t0, const float* t1) {
  #pragma unroll
  for (int l = 0; l < 8; l++) {
    int ix = l * 256 + tid;
    int col = ix & 127;
    int kp  = ix >> 7;
    v2h pk; pk[0] = (h16_t)t0[l]; pk[1] = (h16_t)t1[l];
    *(v2h*)(dst + col * BKSTR + 2 * kp) = pk;   // [col][k] transposed layout
  }
}

__global__ __launch_bounds__(256) void k_gemm(
    const h16_t* __restrict__ A, const float* __restrict__ Bw,
    const float* __restrict__ bias, const float* __restrict__ res,
    float* __restrict__ outF, h16_t* __restrict__ outH,
    int M, int N, int K, int act) {
  __shared__ __align__(16) h16_t sA[2][BM * SA_STR];
  __shared__ __align__(16) h16_t sBt[2][BN * BKSTR];   // [col][k]
  const int tid = threadIdx.x;
  const int lane = tid & 31;
  const int wave = tid >> 5;
  const int wm = wave & 3;            // 4 row waves  (32 rows each)
  const int wn = wave >> 2;           // 2 col waves  (64 cols each)
  const int blockM = blockIdx.y * BM;
  const int blockN = blockIdx.x * BN;

  v8f acc[2][4];
  #pragma unroll
  for (int i = 0; i < 2; i++)
    #pragma unroll
    for (int j = 0; j < 4; j++)
      #pragma unroll
      for (int r = 0; r < 8; r++) acc[i][j][r] = 0.f;

  const int alr   = lane & 15;
  const int kbase = (lane >> 4) * 8;
  const int kb2   = (lane >> 4) * 16;

  float tb0[8], tb1[8];
  const int nk = K / BK;

  // prologue: stage tile 0
  copyA_tile(A, sA[0], tid, blockM, K, 0);
  loadB_regs(Bw, tid, blockN, N, 0, tb0, tb1);
  storeB_lds(sBt[0], tid, tb0, tb1);
#if HAVE_ASYNC_LDS
  wait_async0();
#endif
  __syncthreads();

  for (int kt = 0; kt < nk; kt++) {
    const int cur = kt & 1;
    const bool more = (kt + 1) < nk;
    if (more) {
      copyA_tile(A, sA[cur ^ 1], tid, blockM, K, (kt + 1) * BK);
      loadB_regs(Bw, tid, blockN, N, (kt + 1) * BK, tb0, tb1);
    }

    // A fragments (2 row tiles): ds_load_b64 x4 each
    v16h af[2];
    #pragma unroll
    for (int mi = 0; mi < 2; mi++) {
      const h16_t* ap = sA[cur] + (wm * 32 + mi * 16 + alr) * SA_STR;
      Frag16 fa;
      fa.q[0] = *(const v4h*)(ap + kbase);
      fa.q[1] = *(const v4h*)(ap + kbase + 4);
      fa.q[2] = *(const v4h*)(ap + 16 + kbase);
      fa.q[3] = *(const v4h*)(ap + 16 + kbase + 4);
      af[mi] = fa.v;
    }
    // B fragments (4 col tiles): ds_load_b128 x2 each
    v16h bf[4];
    #pragma unroll
    for (int ni = 0; ni < 4; ni++) {
      const h16_t* bp = sBt[cur] + (wn * 64 + ni * 16 + alr) * BKSTR;
      Frag16 fb;
      fb.h[0] = *(const v8h*)(bp + kbase);
      fb.h[1] = *(const v8h*)(bp + 16 + kbase);
      bf[ni] = fb.v;
    }
    #pragma unroll
    for (int mi = 0; mi < 2; mi++)
      #pragma unroll
      for (int ni = 0; ni < 4; ni++)
        acc[mi][ni] = wmma_f16(af[mi], bf[ni], acc[mi][ni]);

    if (more) {
      storeB_lds(sBt[cur ^ 1], tid, tb0, tb1);
#if HAVE_ASYNC_LDS
      wait_async0();
#endif
      __syncthreads();
    }
  }

  // epilogue
  const int rowoff = (lane >> 4) * 8;
  #pragma unroll
  for (int mi = 0; mi < 2; mi++) {
    #pragma unroll
    for (int ni = 0; ni < 4; ni++) {
      int gcol = blockN + wn * 64 + ni * 16 + alr;
      if (gcol >= N) continue;
      float bv = bias ? bias[gcol] : 0.f;
      #pragma unroll
      for (int r = 0; r < 8; r++) {
        int grow = blockM + wm * 32 + mi * 16 + rowoff + r;
        float v = acc[mi][ni][r] + bv;
        if (res)  v += res[(long long)grow * N + gcol];
        if (act == 1) v = gelu_f(v);
        if (outF) outF[(long long)grow * N + gcol] = v;
        if (outH) outH[(long long)grow * N + gcol] = (h16_t)v;
      }
    }
  }
}

// ---------------- fused causal attention for one (b,h) x 32-row tile -------
// dynamic LDS: float sS[32][SSTR]
#define SSTR 1028            // 4112B row stride: 16B aligned, conflict-free
#define ATT_LDS (32 * SSTR * 4)

__global__ __launch_bounds__(256) void k_attn(const h16_t* __restrict__ q16,
                                              const h16_t* __restrict__ k16,
                                              const h16_t* __restrict__ vT16,
                                              h16_t* __restrict__ y16) {
  extern __shared__ float sS[];
  const int bh = blockIdx.y;                 // b*HH + h
  const int r0 = blockIdx.x * 32;            // query rows [r0, r0+32)
  const int tid = threadIdx.x, lane = tid & 31, wave = tid >> 5;
  const int h = bh % HH;
  const int b = bh / HH;
  const h16_t* qb  = q16  + (size_t)bh * TT * HD;
  const h16_t* kb  = k16  + (size_t)bh * TT * HD;
  const h16_t* vTb = vT16 + (size_t)bh * TT * HD;   // [hd][t]
  const int alr    = lane & 15;
  const int kbase  = (lane >> 4) * 8;
  const int kb2    = (lane >> 4) * 16;
  const int rowoff = (lane >> 4) * 8;

  // Q fragments: 2 row tiles x 2 hd-chunks of 32 (b128 global loads)
  v16h aq[2][2];
  #pragma unroll
  for (int mi = 0; mi < 2; mi++) {
    const h16_t* qr = qb + (size_t)(r0 + mi * 16 + alr) * HD;
    #pragma unroll
    for (int ch = 0; ch < 2; ch++) {
      const h16_t* qc = qr + ch * 32;
      Frag16 fq;
      fq.h[0] = *(const v8h*)(qc + kbase);
      fq.h[1] = *(const v8h*)(qc + 16 + kbase);
      aq[mi][ch] = fq.v;
    }
  }

  // S = Q K^T (scaled + causal mask) -> LDS; this wave's 128-col strip
  #pragma unroll
  for (int ct = 0; ct < 8; ct++) {
    int key = wave * 128 + ct * 16 + alr;            // key column index
    v16h bk[2];
    #pragma unroll
    for (int ch = 0; ch < 2; ch++) {
      const h16_t* kr = kb + (size_t)key * HD + ch * 32;
      Frag16 fk;
      fk.h[0] = *(const v8h*)(kr + kbase);
      fk.h[1] = *(const v8h*)(kr + 16 + kbase);
      bk[ch] = fk.v;
    }
    #pragma unroll
    for (int mi = 0; mi < 2; mi++) {
      v8f acc;
      #pragma unroll
      for (int r = 0; r < 8; r++) acc[r] = 0.f;
      acc = wmma_f16(aq[mi][0], bk[0], acc);
      acc = wmma_f16(aq[mi][1], bk[1], acc);
      #pragma unroll
      for (int r = 0; r < 8; r++) {
        int lrow = mi * 16 + rowoff + r;
        int grow = r0 + lrow;
        int gcol = wave * 128 + ct * 16 + alr;
        float v = acc[r] * 0.125f;                   // 1/sqrt(64)
        if (gcol > grow) v = -1e29f;                 // NEG_INF, matches ref
        sS[lrow * SSTR + gcol] = v;
      }
    }
  }
  __syncthreads();

  // softmax over 1024 cols; each wave owns 4 rows
  #pragma unroll
  for (int rr = 0; rr < 4; rr++) {
    float* row = &sS[(wave * 4 + rr) * SSTR];
    float mx = -3.0e38f;
    for (int j = lane; j < TT; j += 32) mx = fmaxf(mx, row[j]);
    #pragma unroll
    for (int off = 16; off > 0; off >>= 1) mx = fmaxf(mx, __shfl_xor(mx, off, 32));
    float sum = 0.f;
    for (int j = lane; j < TT; j += 32) { float e = __expf(row[j] - mx); row[j] = e; sum += e; }
    #pragma unroll
    for (int off = 16; off > 0; off >>= 1) sum += __shfl_xor(sum, off, 32);
    float inv = 1.0f / sum;
    for (int j = lane; j < TT; j += 32) row[j] *= inv;
  }
  __syncthreads();

  // Y = P @ V : 8 waves -> 2x4 tiles of 16x16 (rows x hd)
  const int mt = wave & 1, nt = wave >> 1;
  v8f acc;
  #pragma unroll
  for (int r = 0; r < 8; r++) acc[r] = 0.f;
  const int hdc = nt * 16 + alr;
  for (int kc = 0; kc < TT / 32; kc++) {
    // P fragment: aligned float4 LDS loads + cvt to f16
    const float* pr = &sS[(mt * 16 + alr) * SSTR + kc * 32];
    v4f f0 = *(const v4f*)(pr + kbase);
    v4f f1 = *(const v4f*)(pr + kbase + 4);
    v4f f2 = *(const v4f*)(pr + 16 + kbase);
    v4f f3 = *(const v4f*)(pr + 16 + kbase + 4);
    Frag16 fp;
    #pragma unroll
    for (int e = 0; e < 4; e++) {
      fp.v[e]      = (h16_t)f0[e];
      fp.v[4 + e]  = (h16_t)f1[e];
      fp.v[8 + e]  = (h16_t)f2[e];
      fp.v[12 + e] = (h16_t)f3[e];
    }
    // V fragment from transposed V: two b128 global loads
    const h16_t* vr = vTb + (size_t)hdc * TT + kc * 32;
    Frag16 fv;
    fv.h[0] = *(const v8h*)(vr + kbase);
    fv.h[1] = *(const v8h*)(vr + 16 + kbase);
    acc = wmma_f16(fp.v, fv.v, acc);
  }
  #pragma unroll
  for (int r = 0; r < 8; r++) {
    int t = r0 + mt * 16 + rowoff + r;
    int col = h * HD + nt * 16 + alr;
    y16[(size_t)(b * TT + t) * CC + col] = (h16_t)acc[r];
  }
}

// ---------------- host orchestration ----------------
extern "C" void kernel_launch(void* const* d_in, const int* in_sizes, int n_in,
                              void* d_out, int out_size, void* d_ws, size_t ws_size,
                              hipStream_t stream) {
  (void)in_sizes; (void)n_in; (void)out_size; (void)ws_size;

  const int*   idx        = (const int*)  d_in[0];
  const float* wte        = (const float*)d_in[1];
  const float* wpe        = (const float*)d_in[2];
  const float* ln1_w      = (const float*)d_in[3];
  const float* ln1_b      = (const float*)d_in[4];
  const float* attn_w     = (const float*)d_in[5];
  const float* attn_b     = (const float*)d_in[6];
  const float* attnproj_w = (const float*)d_in[7];
  const float* attnproj_b = (const float*)d_in[8];
  const float* ln2_w      = (const float*)d_in[9];
  const float* ln2_b      = (const float*)d_in[10];
  const float* fc_w       = (const float*)d_in[11];
  const float* fc_b       = (const float*)d_in[12];
  const float* mlpproj_w  = (const float*)d_in[13];
  const float* mlpproj_b  = (const float*)d_in[14];
  const float* lnf_w      = (const float*)d_in[15];
  const float* lnf_b      = (const float*)d_in[16];
  const float* lmhead_w   = (const float*)d_in[17];

  char* ws = (char*)d_ws;
  size_t off = 0;
  auto carve = [&](size_t bytes) -> void* {
    void* p = ws + off;
    off = (off + bytes + 255) & ~(size_t)255;
    return p;
  };
  float*  xf    = (float*) carve((size_t)MM * CC * 4);
  float*  x2f   = (float*) carve((size_t)MM * CC * 4);
  h16_t*  h16   = (h16_t*) carve((size_t)MM * CC * 2);
  float*  qkvf  = (float*) carve((size_t)MM * 3 * CC * 4);
  h16_t*  q16   = (h16_t*) carve((size_t)MM * CC * 2);
  h16_t*  k16   = (h16_t*) carve((size_t)MM * CC * 2);
  h16_t*  vT16  = (h16_t*) carve((size_t)MM * CC * 2);
  h16_t*  y16   = (h16_t*) carve((size_t)MM * CC * 2);
  h16_t*  act16 = (h16_t*) carve((size_t)MM * 4 * CC * 2);

  // embedding
  k_embed<<<(MM * CC + 255) / 256, 256, 0, stream>>>(idx, wte, wpe, xf);

  for (int l = 0; l < LLAY; l++) {
    // ln1 -> h16
    k_ln<<<MM, 256, 0, stream>>>(xf, ln1_w + l * CC, ln1_b + l * CC, h16);
    // qkv = h16 @ attn_w + attn_b   [2048, 2304]
    k_gemm<<<dim3((3 * CC) / BN, MM / BM), 256, 0, stream>>>(
        h16, attn_w + (size_t)l * CC * 3 * CC, attn_b + (size_t)l * 3 * CC,
        nullptr, qkvf, nullptr, MM, 3 * CC, CC, 0);
    // split heads -> f16 (v transposed)
    k_pack<<<(MM * 3 * CC + 255) / 256, 256, 0, stream>>>(qkvf, q16, k16, vT16);
    // fused attention -> y16 [2048, 768]
    k_attn<<<dim3(TT / 32, BB * HH), 256, ATT_LDS, stream>>>(q16, k16, vT16, y16);
    // x2 = xf + y16 @ attnproj_w + b
    k_gemm<<<dim3(CC / BN, MM / BM), 256, 0, stream>>>(
        y16, attnproj_w + (size_t)l * CC * CC, attnproj_b + (size_t)l * CC,
        xf, x2f, nullptr, MM, CC, CC, 0);
    // ln2 -> h16
    k_ln<<<MM, 256, 0, stream>>>(x2f, ln2_w + l * CC, ln2_b + l * CC, h16);
    // act16 = gelu(h16 @ fc_w + fc_b)   [2048, 3072] f16
    k_gemm<<<dim3((4 * CC) / BN, MM / BM), 256, 0, stream>>>(
        h16, fc_w + (size_t)l * CC * 4 * CC, fc_b + (size_t)l * 4 * CC,
        nullptr, nullptr, act16, MM, 4 * CC, CC, 1);
    // xf = x2f + act16 @ mlpproj_w + b
    k_gemm<<<dim3(CC / BN, MM / BM), 256, 0, stream>>>(
        act16, mlpproj_w + (size_t)l * 4 * CC * CC, mlpproj_b + (size_t)l * CC,
        x2f, xf, nullptr, MM, CC, 4 * CC, 0);
  }

  // final LN -> h16, then logits = h16 @ lmhead_w  [2048, 50257]
  k_ln<<<MM, 256, 0, stream>>>(xf, lnf_w, lnf_b, h16);
  k_gemm<<<dim3((VV + BN - 1) / BN, MM / BM), 256, 0, stream>>>(
      h16, lmhead_w, nullptr, nullptr, (float*)d_out, nullptr, MM, VV, CC, 0);
}
